// ObjectTracker_49263274885505
// MI455X (gfx1250) — compile-verified, matrix operand order checked by reading, not run on hardware
//
#include <hip/hip_runtime.h>
#include <math.h>

// ---------------------------------------------------------------------------
// Object tracker update for MI455X (gfx1250, wave32, WMMA).
// B=32 batches, T=256 tracks, N=256 dets, D=1024 feat, C=80 classes.
//
// kernel 1: per-row reciprocal L2 norms (track + det) -> ws
// kernel 2: cosine-sim GEMM via v_wmma_f32_16x16x32_f16.  The 16x1024 track
//           strip is normalized+converted to f16 in LDS once per block
//           (32KB of the 320KB/WGP), so A fragments are pure ds_load_b128
//           and only the det side converts f32->f16 in the hot loop.
//           Combines with IoU + active mask -> 8MB cost matrix in ws.
// kernel 3: per-batch greedy matching with the whole 256x256 cost matrix
//           resident in LDS (256KB), then output materialization.
// ---------------------------------------------------------------------------

typedef __attribute__((ext_vector_type(16))) _Float16 v16h;
typedef __attribute__((ext_vector_type(8)))  _Float16 v8h;
typedef __attribute__((ext_vector_type(8)))  float    v8f;

#define Bc 32
#define Tt 256
#define Nn 256
#define Dd 1024
#define Cc 80
#define SIM_THRESH 0.7f
#define MAX_AGE_K 10
#define W_SIM 0.7f
#define W_IOU 0.3f

// ---------------------------------------------------------------------------
// IoU of (cx,cy,w,h) boxes, matching the reference formula exactly.
__device__ __forceinline__ float iou_box(float4 a, float4 d) {
    float tx1 = a.x - 0.5f * a.z, ty1 = a.y - 0.5f * a.w;
    float tx2 = a.x + 0.5f * a.z, ty2 = a.y + 0.5f * a.w;
    float dx1 = d.x - 0.5f * d.z, dy1 = d.y - 0.5f * d.w;
    float dx2 = d.x + 0.5f * d.z, dy2 = d.y + 0.5f * d.w;
    float iw = fmaxf(fminf(tx2, dx2) - fmaxf(tx1, dx1), 0.0f);
    float ih = fmaxf(fminf(ty2, dy2) - fmaxf(ty1, dy1), 0.0f);
    float inter = iw * ih;
    float uni = a.z * a.w + d.z * d.w - inter;
    return inter / (uni + 1e-6f);
}

// ---------------------------------------------------------------------------
// Kernel 1: reciprocal row norms.  One wave (32 lanes) per 1024-elem row.
// rows [0, B*T) -> track_memory, rows [B*T, B*T+B*N) -> det_memory.
__global__ void __launch_bounds__(256)
k_rnorm(const float* __restrict__ tm, const float* __restrict__ dm,
        float* __restrict__ rn)
{
    const int row  = blockIdx.x * 8 + (threadIdx.x >> 5);
    const int lane = threadIdx.x & 31;
    const int nTrackRows = Bc * Tt;
    const float* p = (row < nTrackRows) ? (tm + (size_t)row * Dd)
                                        : (dm + (size_t)(row - nTrackRows) * Dd);
    float s = 0.0f;
    for (int j = lane; j < Dd; j += 32) { float x = p[j]; s = fmaf(x, x, s); }
    for (int m = 16; m >= 1; m >>= 1) s += __shfl_xor(s, m, 32);
    if (lane == 0) rn[row] = 1.0f / fmaxf(sqrtf(s), 1e-12f);
}

// ---------------------------------------------------------------------------
// CDNA5 16-bit A/B fragment layout (ISA 7.12.2): lane L<16 holds row M=L
// (B: col N=L), K = {k0..k0+7} in halves 0..7 and {k0+16..k0+23} in halves
// 8..15; lanes 16..31 hold the +8 K offset for row M=L-16.
//
// Global f32 path (used for the det/B side): load, scale by rnorm, cvt f16.
__device__ __forceinline__ v16h load_frag_global(const float* __restrict__ base,
                                                 int row0, int k0, int lane, float s)
{
    const int r   = lane & 15;
    const int sel = (lane >> 4) << 3;
    const float* p = base + (size_t)(row0 + r) * Dd + k0 + sel;
    const float4* q0 = (const float4*)p;          // 32B-aligned offsets
    const float4* q1 = (const float4*)(p + 16);
    float4 x0 = q0[0], x1 = q0[1];
    float4 y0 = q1[0], y1 = q1[1];
    v16h f;
    f[0]  = (_Float16)(x0.x * s); f[1]  = (_Float16)(x0.y * s);
    f[2]  = (_Float16)(x0.z * s); f[3]  = (_Float16)(x0.w * s);
    f[4]  = (_Float16)(x1.x * s); f[5]  = (_Float16)(x1.y * s);
    f[6]  = (_Float16)(x1.z * s); f[7]  = (_Float16)(x1.w * s);
    f[8]  = (_Float16)(y0.x * s); f[9]  = (_Float16)(y0.y * s);
    f[10] = (_Float16)(y0.z * s); f[11] = (_Float16)(y0.w * s);
    f[12] = (_Float16)(y1.x * s); f[13] = (_Float16)(y1.y * s);
    f[14] = (_Float16)(y1.z * s); f[15] = (_Float16)(y1.w * s);
    return f;
}

// ---------------------------------------------------------------------------
// Kernel 2: cost matrix.  grid (T/16, B); 512 threads = 16 waves.
// blockIdx.x = ti (16-row track block), wave id = tj (16-col det block).
// A strip staged in LDS as normalized f16 (32KB) once per block.
__global__ void __launch_bounds__(512)
k_gemm(const float* __restrict__ tm, const float* __restrict__ dm,
       const float* __restrict__ rnT, const float* __restrict__ rnD,
       const float* __restrict__ tbx, const float* __restrict__ dbx,
       const unsigned char* __restrict__ tact, float* __restrict__ Cws)
{
    __shared__ _Float16 Alds[16 * Dd];    // 32 KB: 16 track rows, f16, normalized

    const int ti   = blockIdx.x;          // 0..15
    const int b    = blockIdx.y;          // 0..31
    const int tid  = threadIdx.x;
    const int lane = tid & 31;
    const int tj   = tid >> 5;            // 0..15

    const float* tmB = tm + ((size_t)b * Tt + (size_t)ti * 16) * Dd;
    const float* dmB = dm + (size_t)b * Nn * Dd;

    // ---- cooperative stage: normalize + convert the A strip into LDS ----
    for (int i = tid; i < (16 * Dd) / 4; i += 512) {
        const int row = i >> 8;                       // Dd/4 = 256 float4 per row
        const float s = rnT[b * Tt + ti * 16 + row];
        float4 x = ((const float4*)tmB)[i];
        _Float16* o = &Alds[i * 4];
        o[0] = (_Float16)(x.x * s); o[1] = (_Float16)(x.y * s);
        o[2] = (_Float16)(x.z * s); o[3] = (_Float16)(x.w * s);
    }
    __syncthreads();

    const float sB = rnD[b * Nn + tj * 16 + (lane & 15)];
    const int r    = lane & 15;
    const int sel  = (lane >> 4) << 3;
    const _Float16* arow = &Alds[r * Dd + sel];

    v8f acc = {};
    for (int k0 = 0; k0 < Dd; k0 += 32) {
        // A fragment: two ds_load_b128 + register concat, zero ALU
        v8h a0 = *(const v8h*)(arow + k0);
        v8h a1 = *(const v8h*)(arow + k0 + 16);
        v16h a = __builtin_shufflevector(a0, a1, 0, 1, 2, 3, 4, 5, 6, 7,
                                                 8, 9, 10, 11, 12, 13, 14, 15);
        v16h bb = load_frag_global(dmB, tj * 16, k0, lane, sB);
        acc = __builtin_amdgcn_wmma_f32_16x16x32_f16(
            /*neg_a=*/false, a, /*neg_b=*/false, bb,
            /*c_mod=*/(short)0, acc, /*reuse_a=*/false, /*reuse_b=*/false);
    }

    // C/D layout: lane<16 -> N=lane, M=vgpr; lane>=16 -> N=lane-16, M=vgpr+8.
    const int n     = tj * 16 + (lane & 15);
    const int mbase = ti * 16 + ((lane >> 4) << 3);
    const float4 dbv = ((const float4*)dbx)[(size_t)b * Nn + n];
    float* Cb = Cws + (size_t)b * Tt * Nn;
#pragma unroll
    for (int i = 0; i < 8; ++i) {
        int t = mbase + i;
        float sc = -1.0f;
        if (tact[(size_t)b * Tt + t]) {
            float4 tbv = ((const float4*)tbx)[(size_t)b * Tt + t];
            sc = W_SIM * acc[i] + W_IOU * iou_box(tbv, dbv);
        }
        Cb[(size_t)t * Nn + n] = sc;
    }
}

// ---------------------------------------------------------------------------
// Kernel 3: greedy matching + output materialization.  One block per batch.
struct SMem3 {
    float  Cmat[Tt * Nn];   // 256 KB cost matrix, fully resident in LDS
    float4 tb[Tt];
    float4 db[Nn];
    float  rowMax[Tt];
    int    rowArg[Tt];
    float  redV[256];
    int    redI[256];
    int    matchDet[Tt];
    int    act[Tt];
    int    curD;
    int    doneFlag;
};

__global__ void __launch_bounds__(512)
k_match(const float* __restrict__ Cws,
        const float* __restrict__ tm,  const float* __restrict__ dm,
        const float* __restrict__ tbx, const float* __restrict__ dbx,
        const unsigned char* __restrict__ tact,
        const int* __restrict__ tage, const int* __restrict__ thits,
        const float* __restrict__ logits,
        float* __restrict__ out)
{
    extern __shared__ __align__(16) char smem_raw[];
    SMem3& sm = *reinterpret_cast<SMem3*>(smem_raw);
    const int b   = blockIdx.x;
    const int tid = threadIdx.x;

    // ---- stage cost matrix + metadata into LDS ----
    {
        const float4* src = (const float4*)(Cws + (size_t)b * Tt * Nn);
        float4* dst = (float4*)sm.Cmat;
        for (int i = tid; i < (Tt * Nn) / 4; i += 512) dst[i] = src[i];
    }
    if (tid < Tt) {
        sm.tb[tid] = ((const float4*)tbx)[(size_t)b * Tt + tid];
        sm.db[tid] = ((const float4*)dbx)[(size_t)b * Nn + tid];
        sm.act[tid] = tact[(size_t)b * Tt + tid] ? 1 : 0;
        sm.matchDet[tid] = -1;
    }
    if (tid == 0) sm.doneFlag = 0;
    __syncthreads();

    // ---- per-row max cache ----
    if (tid < Tt) {
        float best = -INFINITY; int barg = 0;
        const float* row = &sm.Cmat[tid * Nn];
        for (int n = 0; n < Nn; ++n) {
            float v = row[n];
            if (v > best) { best = v; barg = n; }   // first max (ties -> lowest n)
        }
        sm.rowMax[tid] = best; sm.rowArg[tid] = barg;
    }
    __syncthreads();

    // ---- sequential greedy matching (monotone non-increasing max => early exit) ----
    for (int step = 0; step < Tt; ++step) {
        if (tid < 256) { sm.redV[tid] = sm.rowMax[tid]; sm.redI[tid] = tid; }
        __syncthreads();
        for (int s = 128; s >= 1; s >>= 1) {
            if (tid < s) {
                float va = sm.redV[tid], vb = sm.redV[tid + s];
                int   ia = sm.redI[tid], ib = sm.redI[tid + s];
                // ties -> lowest track index (matches flat argmax semantics)
                if (vb > va || (vb == va && ib < ia)) { sm.redV[tid] = vb; sm.redI[tid] = ib; }
            }
            __syncthreads();
        }
        if (tid == 0) {
            float val = sm.redV[0];
            if (val < SIM_THRESH) {
                sm.doneFlag = 1;
            } else {
                int t = sm.redI[0];
                int d = sm.rowArg[t];
                sm.matchDet[t] = d;
                sm.rowMax[t]   = -INFINITY;
                sm.curD        = d;
            }
        }
        __syncthreads();
        if (sm.doneFlag) break;
        const int d = sm.curD;
        if (tid < Tt) sm.Cmat[tid * Nn + d] = -1.0f;    // remove column d
        __syncthreads();
        if (tid < Tt && sm.matchDet[tid] < 0 && sm.rowArg[tid] == d) {
            float best = -INFINITY; int barg = 0;
            const float* row = &sm.Cmat[tid * Nn];
            for (int n = 0; n < Nn; ++n) {
                float v = row[n];
                if (v > best) { best = v; barg = n; }
            }
            sm.rowMax[tid] = best; sm.rowArg[tid] = barg;
        }
        __syncthreads();
    }
    __syncthreads();

    // ---- outputs (all written as f32, concatenated in reference return order) ----
    float* out_mem   = out;
    float* out_box   = out_mem  + (size_t)Bc * Tt * Dd;
    float* out_act   = out_box  + (size_t)Bc * Tt * 4;
    float* out_age   = out_act  + (size_t)Bc * Tt;
    float* out_hits  = out_age  + (size_t)Bc * Tt;
    float* out_match = out_hits + (size_t)Bc * Tt;
    float* out_score = out_match + (size_t)Bc * Tt;

    const float* tmB = tm + (size_t)b * Tt * Dd;
    const float* dmB = dm + (size_t)b * Nn * Dd;

    // new_mem: matched tracks take det feature row, else keep track row (float4 copies)
    {
        float4* omem = (float4*)(out_mem + (size_t)b * Tt * Dd);
        for (int i = tid; i < (Tt * Dd) / 4; i += 512) {
            int t  = i >> 8;            // 256 float4 per 1024-elem row
            int c4 = i & 255;
            int d  = sm.matchDet[t];
            const float4* src = (d >= 0) ? (const float4*)(dmB + (size_t)d * Dd)
                                         : (const float4*)(tmB + (size_t)t * Dd);
            omem[i] = src[c4];
        }
    }

    if (tid < Tt) {
        const int md = sm.matchDet[tid];
        const int matched = (md >= 0);
        // new_box
        float4 bx = matched ? sm.db[md] : sm.tb[tid];
        float* ob = out_box + ((size_t)b * Tt + tid) * 4;
        ob[0] = bx.x; ob[1] = bx.y; ob[2] = bx.z; ob[3] = bx.w;
        // scalar state updates
        const size_t o1 = (size_t)b * Tt + tid;
        int age    = tage[o1];
        int hits   = thits[o1];
        int active = sm.act[tid];
        int nhits  = hits + (matched ? 1 : 0);
        int ua     = active && !matched;
        int nage   = matched ? 0 : (ua ? age + 1 : age);
        int nact   = matched ? 1 : (ua ? ((nage <= MAX_AGE_K) ? 1 : 0) : active);
        out_act[o1]   = (float)nact;
        out_age[o1]   = (float)nage;
        out_hits[o1]  = (float)nhits;
        out_match[o1] = (float)md;
    }

    // scores: max(softmax(logits)) = 1 / sum(exp(l - max(l)))
    if (tid < Nn) {
        const float* lg = logits + ((size_t)b * Nn + tid) * Cc;
        float mx = -INFINITY;
        for (int c = 0; c < Cc; ++c) mx = fmaxf(mx, lg[c]);
        float se = 0.0f;
        for (int c = 0; c < Cc; ++c) se += expf(lg[c] - mx);
        out_score[(size_t)b * Nn + tid] = 1.0f / se;
    }
}

// ---------------------------------------------------------------------------
extern "C" void kernel_launch(void* const* d_in, const int* in_sizes, int n_in,
                              void* d_out, int out_size, void* d_ws, size_t ws_size,
                              hipStream_t stream)
{
    const float*         tm   = (const float*)d_in[0];          // [B,T,D] f32
    const float*         tbx  = (const float*)d_in[1];          // [B,T,4] f32
    const unsigned char* tact = (const unsigned char*)d_in[2];  // [B,T] bool (1B)
    const int*           tage = (const int*)d_in[3];            // [B,T] i32
    const int*           thit = (const int*)d_in[4];            // [B,T] i32
    const float*         dbx  = (const float*)d_in[5];          // [B,N,4] f32
    const float*         dm   = (const float*)d_in[6];          // [B,N,D] f32
    const float*         lg   = (const float*)d_in[7];          // [B,N,C] f32
    float* out = (float*)d_out;

    // workspace: rnorms (B*(T+N) f32 = 64KB) then cost matrix (B*T*N f32 = 8MB)
    float* rn  = (float*)d_ws;
    float* Cws = rn + (size_t)Bc * (Tt + Nn);

    k_rnorm<<<dim3((Bc * (Tt + Nn)) / 8), dim3(256), 0, stream>>>(tm, dm, rn);
    k_gemm <<<dim3(Tt / 16, Bc), dim3(512), 0, stream>>>(
        tm, dm, rn, rn + (size_t)Bc * Tt, tbx, dbx, tact, Cws);
    k_match<<<dim3(Bc), dim3(512), sizeof(SMem3), stream>>>(
        Cws, tm, dm, tbx, dbx, tact, tage, thit, lg, out);
}
